// DLRM_10342281249356
// MI455X (gfx1250) — matrix operand order repaired and checked by reading, not compiled, not run on hardware
//
#include <hip/hip_runtime.h>

typedef __attribute__((ext_vector_type(16))) _Float16     v16h;
typedef __attribute__((ext_vector_type(8)))  _Float16     v8h;
typedef __attribute__((ext_vector_type(4)))  _Float16     v4h;
typedef __attribute__((ext_vector_type(8)))  float        v8f;
typedef __attribute__((ext_vector_type(4)))  float        v4f;
typedef __attribute__((ext_vector_type(4)))  unsigned int v4u;
typedef __attribute__((ext_vector_type(8)))  int          v8i;
typedef __attribute__((ext_vector_type(4)))  int          v4i;

#define BATCH   16384
#define KTAB    26
#define DDIM    128
#define NCOLL_  1000
#define QROWS_  1000
#define ZW      480   // 128 dense + 351 triu + 1 pad (weight-padded with zeros)

// ---------------------------------------------------------------------------
// WMMA fragment loaders (per CDNA5 ISA 16-bit layouts, wave32)
// A 16x32: lane lo holds row M=lo; K = {hi*8..hi*8+7} and {16+hi*8..+7}
// B 32x16 (sourced from row-major W^T / T): lane lo holds col N=lo;
//          K = {hi*8..hi*8+7} and {16+hi*8..+7} (same order as A)
// C 16x16: vgpr r <-> M = r + 8*hi, N = lo
// ---------------------------------------------------------------------------
__device__ __forceinline__ v16h load_a_frag(const _Float16* p, int hi) {
    v8h c0 = *(const v8h*)(p + hi * 8);
    v8h c1 = *(const v8h*)(p + 16 + hi * 8);
    v16h f;
#pragma unroll
    for (int t = 0; t < 8; ++t) { f[t] = c0[t]; f[t + 8] = c1[t]; }
    return f;
}

__device__ __forceinline__ v16h load_b_frag(const _Float16* p, int hi) {
    v8h c0 = *(const v8h*)(p + hi * 8);
    v8h c1 = *(const v8h*)(p + 16 + hi * 8);
    v16h f;
#pragma unroll
    for (int t = 0; t < 8; ++t) { f[t] = c0[t]; f[t + 8] = c1[t]; }
    return f;
}

// ---------------------------------------------------------------------------
// Tensor Data Mover: stage a contiguous slab (global -> LDS) with one D#.
// 1-D tile of 8-byte elements; groups 2/3 zeroed (<=2D tensor).
// D# packing per CDNA5 ISA ch.8 (group0: count/lds/global/type, group1: dims).
// 6-arg builtin form (clang-23 / therock-10.0 headers).
// ---------------------------------------------------------------------------
__device__ __forceinline__ void tdm_load_slab(const _Float16* gsrc,
                                              _Float16* lds_dst,
                                              unsigned int nbytes)
{
    unsigned long long ga = (unsigned long long)(uintptr_t)gsrc;
    unsigned int la       = (unsigned int)(uintptr_t)lds_dst;   // low 32b = LDS offset
    unsigned int nelem    = nbytes >> 3;                        // 8-byte elements

    v4u g0;
    g0[0] = 1u;                                                 // count=1 (valid), user mode
    g0[1] = la;                                                 // lds_addr (bytes)
    g0[2] = (unsigned int)ga;                                   // global_addr[31:0]
    g0[3] = (unsigned int)((ga >> 32) & 0x1FFFFFFull)           // global_addr[56:32]
          | (2u << 30);                                         // type = 2 ("image")

    v8i g1;
    g1[0] = (int)(3u << 16);                                    // wg_mask=0, data_size=3 (8B)
    g1[1] = (int)((nelem & 0xFFFFu) << 16);                     // tensor_dim0[15:0]
    g1[2] = (int)(((nelem >> 16) & 0xFFFFu) | (1u << 16));      // tensor_dim0[31:16] | tensor_dim1=1
    g1[3] = (int)((nelem & 0xFFFFu) << 16);                     // tensor_dim1[31:16]=0 | tile_dim0=nelem
    g1[4] = 0;                                                  // tile_dim1=0 (1-D), tile_dim2=0
    g1[5] = (int)nelem;                                         // tensor_dim0_stride[31:0]
    g1[6] = 0;                                                  // stride[47:32]=0, dim1_stride lo=0
    g1[7] = 0;

    v4i z4 = {0, 0, 0, 0};
    v8i z8 = {0, 0, 0, 0, 0, 0, 0, 0};
    __builtin_amdgcn_tensor_load_to_lds(g0, g1, z4, z4, z8, 0);
}

// ---------------------------------------------------------------------------
// f16 WMMA GEMM:  C[M,N] = relu?( A[M,K] @ W + bias )  (f32 accum)
// A: row-major f16, stride lda.  Wt: W^T, N x K row-major f16.  C: f16, stride N.
// Block = 8 waves covering 256 M-rows x 32 N-cols. The block's 32 x K weight
// slab (<=64KB) is staged into LDS once via the Tensor Data Mover; each wave
// then computes a 32x32 tile (2x2 WMMA subtiles), A streamed from global.
// M mult of 256, N mult of 32, K mult of 32 and <= 1024.
// ---------------------------------------------------------------------------
__global__ void gemm_wmma_f16(const _Float16* __restrict__ A,
                              const _Float16* __restrict__ Wt,
                              const float*    __restrict__ bias,
                              _Float16*       __restrict__ C,
                              int M, int N, int K, int lda, int relu)
{
    __shared__ _Float16 wsm[32 * 1024];      // 64 KB weight slab

    int tn   = N >> 5;
    int bm   = blockIdx.x / tn;
    int bn   = blockIdx.x - bm * tn;
    int ww   = threadIdx.x >> 5;
    int lane = threadIdx.x & 31;
    int lo   = lane & 15;
    int hi   = lane >> 4;

    if (threadIdx.x < 32) {                  // wave 0 programs the TDM
        tdm_load_slab(Wt + (size_t)(bn * 32) * K, wsm, (unsigned int)(32 * K * 2));
        __builtin_amdgcn_s_wait_tensorcnt(0);
    }
    __syncthreads();

    int wm = bm * 8 + ww;
    const _Float16* Arow0 = A + (size_t)(wm * 32 + lo) * lda;
    const _Float16* Arow1 = Arow0 + (size_t)16 * lda;
    const _Float16* Brow0 = wsm + (size_t)lo * K;
    const _Float16* Brow1 = Brow0 + (size_t)16 * K;

    v8f acc00 = {}, acc01 = {}, acc10 = {}, acc11 = {};

    for (int kk = 0; kk < K; kk += 32) {
        v16h a0 = load_a_frag(Arow0 + kk, hi);
        v16h a1 = load_a_frag(Arow1 + kk, hi);
        v16h b0 = load_b_frag(Brow0 + kk, hi);   // ds_load path
        v16h b1 = load_b_frag(Brow1 + kk, hi);
        acc00 = __builtin_amdgcn_wmma_f32_16x16x32_f16(false, a0, false, b0, (short)0, acc00, false, false);
        acc01 = __builtin_amdgcn_wmma_f32_16x16x32_f16(false, a0, false, b1, (short)0, acc01, false, false);
        acc10 = __builtin_amdgcn_wmma_f32_16x16x32_f16(false, a1, false, b0, (short)0, acc10, false, false);
        acc11 = __builtin_amdgcn_wmma_f32_16x16x32_f16(false, a1, false, b1, (short)0, acc11, false, false);
    }

    v8f* accs[2][2] = { { &acc00, &acc01 }, { &acc10, &acc11 } };
#pragma unroll
    for (int i = 0; i < 2; ++i) {
#pragma unroll
        for (int j = 0; j < 2; ++j) {
            int   n  = bn * 32 + 16 * j + lo;
            float bv = bias[n];
#pragma unroll
            for (int r = 0; r < 8; ++r) {
                int   m = wm * 32 + 16 * i + r + 8 * hi;
                float v = (*accs[i][j])[r] + bv;
                if (relu) v = v > 0.f ? v : 0.f;
                C[(size_t)m * N + n] = (_Float16)v;
            }
        }
    }
}

// ---------------------------------------------------------------------------
// Fused sparse-embedding + pairwise interaction.
// One wave per sample b: build T (32 x 128 f16, rows 27..31 zero) in LDS,
// then Z = T @ T^T via 16 WMMAs, scatter upper triangle + dense into z.
// Block = 128 threads = 4 waves, 32 KB LDS.
// ---------------------------------------------------------------------------
__global__ void sparse_interact(const float* __restrict__ eq,
                                const float* __restrict__ er,
                                const int*   __restrict__ idx,
                                const _Float16* __restrict__ dense,
                                _Float16* __restrict__ z)
{
    __shared__ _Float16 sm[4][32 * 128];
    int wslot = threadIdx.x >> 5;
    int b     = blockIdx.x * 4 + wslot;
    int lane  = threadIdx.x & 31;
    int lo    = lane & 15;
    int hi    = lane >> 4;
    _Float16* T = sm[wslot];
    int c0 = lane * 4;

    // row 0: dense (bottom-MLP output)
    *(v4h*)(T + c0) = *(const v4h*)(dense + (size_t)b * DDIM + c0);

    // rows 1..26: QR-trick embedding products (bag size is 1 in this workload)
#pragma unroll 1
    for (int k = 0; k < KTAB; ++k) {
        int id = idx[k * BATCH + b];
        int qi = id / NCOLL_;
        int ri = id - qi * NCOLL_;
        v4f qv = *(const v4f*)(eq + ((size_t)k * QROWS_ + qi) * DDIM + c0);
        v4f rv = *(const v4f*)(er + ((size_t)k * NCOLL_ + ri) * DDIM + c0);
        v4h pv;
#pragma unroll
        for (int t = 0; t < 4; ++t) pv[t] = (_Float16)(qv[t] * rv[t]);
        *(v4h*)(T + (size_t)(k + 1) * DDIM + c0) = pv;
    }
    // rows 27..31: zero padding
#pragma unroll
    for (int row = 27; row < 32; ++row) {
        v4h zv = {};
        *(v4h*)(T + (size_t)row * DDIM + c0) = zv;
    }
    __syncthreads();

    // Z = T @ T^T  (32x32, K=128): B-fragment of T^T == row-major rows of T
    v8f acc00 = {}, acc01 = {}, acc10 = {}, acc11 = {};
    for (int kk = 0; kk < DDIM; kk += 32) {
        v16h a0 = load_a_frag(T + (size_t)(0  + lo) * DDIM + kk, hi);
        v16h a1 = load_a_frag(T + (size_t)(16 + lo) * DDIM + kk, hi);
        v16h b0 = load_b_frag(T + (size_t)(0  + lo) * DDIM + kk, hi);
        v16h b1 = load_b_frag(T + (size_t)(16 + lo) * DDIM + kk, hi);
        acc00 = __builtin_amdgcn_wmma_f32_16x16x32_f16(false, a0, false, b0, (short)0, acc00, false, false);
        acc01 = __builtin_amdgcn_wmma_f32_16x16x32_f16(false, a0, false, b1, (short)0, acc01, false, false);
        acc10 = __builtin_amdgcn_wmma_f32_16x16x32_f16(false, a1, false, b0, (short)0, acc10, false, false);
        acc11 = __builtin_amdgcn_wmma_f32_16x16x32_f16(false, a1, false, b1, (short)0, acc11, false, false);
    }

    _Float16* zb = z + (size_t)b * ZW;
    // dense passthrough: z[:, 0:128]
    *(v4h*)(zb + c0) = *(const v4h*)(T + c0);

    // upper triangle (m < n <= 26), np.triu_indices(27, 1) row-major order
    v8f* accs[2][2] = { { &acc00, &acc01 }, { &acc10, &acc11 } };
#pragma unroll
    for (int i = 0; i < 2; ++i) {
#pragma unroll
        for (int j = 0; j < 2; ++j) {
#pragma unroll
            for (int r = 0; r < 8; ++r) {
                int m = 16 * i + r + 8 * hi;
                int n = 16 * j + lo;
                if (n > m && n < 27) {
                    int p = m * 26 - (m * (m - 1)) / 2 + (n - m - 1);
                    zb[DDIM + p] = (_Float16)((*accs[i][j])[r]);
                }
            }
        }
    }
}

// ---------------------------------------------------------------------------
// Final layer: out[b] = dot(a[b, 0:256], w) + bias   (fp32, no relu)
// One wave per sample.
// ---------------------------------------------------------------------------
__global__ void final_dot(const _Float16* __restrict__ a,
                          const float*    __restrict__ w,
                          const float*    __restrict__ bias,
                          float*          __restrict__ out)
{
    int gw   = blockIdx.x * (blockDim.x >> 5) + (threadIdx.x >> 5);
    int lane = threadIdx.x & 31;
    v8h av = *(const v8h*)(a + (size_t)gw * 256 + lane * 8);
    float s = 0.f;
#pragma unroll
    for (int t = 0; t < 8; ++t) s += (float)av[t] * w[lane * 8 + t];
#pragma unroll
    for (int off = 16; off > 0; off >>= 1) s += __shfl_xor(s, off, 32);
    if (lane == 0) out[gw] = s + bias[0];
}

// ---------------------------------------------------------------------------
// Conversion kernels
// ---------------------------------------------------------------------------
__global__ void cvt_pad_x(const float* __restrict__ x, _Float16* __restrict__ xp)
{
    int i = blockIdx.x * blockDim.x + threadIdx.x;   // over BATCH*32
    int b = i >> 5, c = i & 31;
    xp[i] = (c < 13) ? (_Float16)x[b * 13 + c] : (_Float16)0.f;
}

// W (K x N fp32, row-major) -> Wt (N x Kp f16, zero-padded K)
__global__ void cvt_w(const float* __restrict__ w, _Float16* __restrict__ wt,
                      int K, int N, int Kp)
{
    int i = blockIdx.x * blockDim.x + threadIdx.x;
    if (i >= N * Kp) return;
    int n = i / Kp, k = i - n * Kp;
    wt[i] = (k < K) ? (_Float16)w[(size_t)k * N + n] : (_Float16)0.f;
}

// ---------------------------------------------------------------------------
// Host side
// ---------------------------------------------------------------------------
static inline void launch_gemm(const _Float16* A, const _Float16* Wt, const float* bias,
                               _Float16* C, int M, int N, int K, int lda, int relu,
                               hipStream_t stream)
{
    int blocks = (M / 256) * (N / 32);          // 8 waves (256 thr) per block
    gemm_wmma_f16<<<blocks, 256, 0, stream>>>(A, Wt, bias, C, M, N, K, lda, relu);
}

extern "C" void kernel_launch(void* const* d_in, const int* in_sizes, int n_in,
                              void* d_out, int out_size, void* d_ws, size_t ws_size,
                              hipStream_t stream)
{
    const float* x    = (const float*)d_in[0];
    const int*   eidx = (const int*)  d_in[1];
    // d_in[2] = emb_offsets: arange(B) per table -> bags of size 1, unused
    const float* eq   = (const float*)d_in[3];
    const float* er   = (const float*)d_in[4];
    const float* bW[3] = { (const float*)d_in[5],  (const float*)d_in[6],  (const float*)d_in[7]  };
    const float* bb[3] = { (const float*)d_in[8],  (const float*)d_in[9],  (const float*)d_in[10] };
    const float* tW[5] = { (const float*)d_in[11], (const float*)d_in[12], (const float*)d_in[13],
                           (const float*)d_in[14], (const float*)d_in[15] };
    const float* tb[5] = { (const float*)d_in[16], (const float*)d_in[17], (const float*)d_in[18],
                           (const float*)d_in[19], (const float*)d_in[20] };
    float* out = (float*)d_out;

    // bump allocator over workspace
    char* wp = (char*)d_ws;
    auto alloc = [&](size_t bytes) -> void* {
        void* r = (void*)wp;
        wp += (bytes + 255) & ~(size_t)255;
        return r;
    };
    _Float16* Xp    = (_Float16*)alloc((size_t)BATCH * 32   * 2);
    _Float16* a1    = (_Float16*)alloc((size_t)BATCH * 512  * 2);
    _Float16* a2    = (_Float16*)alloc((size_t)BATCH * 256  * 2);
    _Float16* a3    = (_Float16*)alloc((size_t)BATCH * 128  * 2);   // dense
    _Float16* zbuf  = (_Float16*)alloc((size_t)BATCH * ZW   * 2);
    _Float16* ta1   = (_Float16*)alloc((size_t)BATCH * 1024 * 2);
    _Float16* ta2   = (_Float16*)alloc((size_t)BATCH * 1024 * 2);
    _Float16* ta3   = (_Float16*)alloc((size_t)BATCH * 512  * 2);
    _Float16* ta4   = (_Float16*)alloc((size_t)BATCH * 256  * 2);
    _Float16* Wb0   = (_Float16*)alloc((size_t)512  * 32   * 2);
    _Float16* Wb1   = (_Float16*)alloc((size_t)256  * 512  * 2);
    _Float16* Wb2   = (_Float16*)alloc((size_t)128  * 256  * 2);
    _Float16* Wt0   = (_Float16*)alloc((size_t)1024 * 480  * 2);
    _Float16* Wt1   = (_Float16*)alloc((size_t)1024 * 1024 * 2);
    _Float16* Wt2   = (_Float16*)alloc((size_t)512  * 1024 * 2);
    _Float16* Wt3   = (_Float16*)alloc((size_t)256  * 512  * 2);

    // --- conversions ---
    {
        int n = BATCH * 32;
        cvt_pad_x<<<(n + 255) / 256, 256, 0, stream>>>(x, Xp);
    }
    struct { const float* src; _Float16* dst; int K, N, Kp; } wl[7] = {
        { bW[0], Wb0, 13,   512,  32   },
        { bW[1], Wb1, 512,  256,  512  },
        { bW[2], Wb2, 256,  128,  256  },
        { tW[0], Wt0, 479,  1024, 480  },
        { tW[1], Wt1, 1024, 1024, 1024 },
        { tW[2], Wt2, 1024, 512,  1024 },
        { tW[3], Wt3, 512,  256,  512  },
    };
    for (int i = 0; i < 7; ++i) {
        int n = wl[i].N * wl[i].Kp;
        cvt_w<<<(n + 255) / 256, 256, 0, stream>>>(wl[i].src, wl[i].dst,
                                                   wl[i].K, wl[i].N, wl[i].Kp);
    }

    // --- bottom MLP (ReLU on all layers) ---
    launch_gemm(Xp, Wb0, bb[0], a1, BATCH, 512, 32,  32,  1, stream);
    launch_gemm(a1, Wb1, bb[1], a2, BATCH, 256, 512, 512, 1, stream);
    launch_gemm(a2, Wb2, bb[2], a3, BATCH, 128, 256, 256, 1, stream);

    // --- embeddings + pairwise interaction -> z (B x 480) ---
    sparse_interact<<<BATCH / 4, 128, 0, stream>>>(eq, er, eidx, a3, zbuf);

    // --- top MLP (ReLU on all but last) ---
    launch_gemm(zbuf, Wt0, tb[0], ta1, BATCH, 1024, 480,  480,  1, stream);
    launch_gemm(ta1,  Wt1, tb[1], ta2, BATCH, 1024, 1024, 1024, 1, stream);
    launch_gemm(ta2,  Wt2, tb[2], ta3, BATCH, 512,  1024, 1024, 1, stream);
    launch_gemm(ta3,  Wt3, tb[3], ta4, BATCH, 256,  512,  512,  1, stream);

    // --- final 256 -> 1 (fp32 dot, no relu) ---
    final_dot<<<BATCH / 8, 256, 0, stream>>>(ta4, tW[4], tb[4], out);
}
